// LSTMEncoder_37331855737089
// MI455X (gfx1250) — matrix operand order repaired and checked by reading, not compiled, not run on hardware
//
#include <hip/hip_runtime.h>

// ---------------------------------------------------------------------------
// LSTM encoder for MI455X (gfx1250, wave32, WMMA + Tensor Data Mover).
// GEMMs in bf16 via v_wmma_f32_16x16x32_bf16 (f32 accumulate); gates/state f32.
// A-tiles / h-state staged into LDS with tensor_load_to_lds (TDM); an inline-asm
// escape ("lds_publish") keeps the compiler from folding loads of the
// DMA-written LDS (which it otherwise sees as never-stored -> undef).
// ---------------------------------------------------------------------------

typedef __attribute__((ext_vector_type(16))) __bf16 v16bf;
typedef __attribute__((ext_vector_type(8)))  float  v8f;
typedef __attribute__((ext_vector_type(4)))  unsigned int u32x4;
typedef __attribute__((ext_vector_type(8)))  int  i32x8;
typedef __attribute__((ext_vector_type(4)))  int  i32x4;

#define BATCH 32
#define TT    256
#define HH    1024
#define G4H   4096
#define ROWS  8192                  // B*T
#define CS_OFF (8388608)            // B*T*H
#define HS_OFF (8388608 + 65536)    // + L*B*H

#if defined(__gfx1250__) && __has_builtin(__builtin_amdgcn_tensor_load_to_lds) && \
    __has_builtin(__builtin_amdgcn_s_wait_tensorcnt)
#define USE_TDM 1
#else
#define USE_TDM 0
#endif

union ABf { v16bf v; uint4 q[2]; unsigned short s[16]; };

__device__ __forceinline__ unsigned short f2bf(float f) {
    union { float f; unsigned int u; } v; v.f = f;
    unsigned int u = v.u;
    unsigned int r = u + 0x7FFFu + ((u >> 16) & 1u);   // round-to-nearest-even
    return (unsigned short)(r >> 16);
}
__device__ __forceinline__ float bf2f(unsigned short s) {
    union { unsigned int u; float f; } v; v.u = ((unsigned int)s) << 16; return v.f;
}
__device__ __forceinline__ float sigmoidf(float x) { return 1.0f / (1.0f + expf(-x)); }

// Make an LDS buffer's contents "visible" to the optimizer after a TDM write:
// the pointer escapes into an asm that may write memory, so subsequent
// ds_loads cannot be folded to undef / forwarded across it.
__device__ __forceinline__ void lds_publish(const void* p) {
    void* q = (void*)p;
    asm volatile("" : "+v"(q) :: "memory");
}

#if USE_TDM
// Issue a 2D TDM load: tile_d1 rows x tile_d0 elems (2-byte elems), row stride
// stride_elems, from gaddr into LDS byte offset lds_off (rows packed tight).
__device__ __forceinline__ void tdm_load_2d(unsigned lds_off, const void* gaddr,
                                            unsigned tile_d0, unsigned tile_d1,
                                            unsigned tensor_d0, unsigned tensor_d1,
                                            unsigned stride_elems)
{
    unsigned long long ga = (unsigned long long)gaddr;
    u32x4 g0;
    g0[0] = 1u;                                             // count=1 (valid), flags 0
    g0[1] = lds_off;                                        // lds_addr
    g0[2] = (unsigned)(ga & 0xFFFFFFFFu);                   // global_addr[31:0]
    g0[3] = (unsigned)((ga >> 32) & 0x01FFFFFFu) | (2u << 30);  // addr[56:32] | type=2
    i32x8 g1;
    g1[0] = (int)(1u << 16);                                // wg_mask=0, data_size=1 (2B)
    g1[1] = (int)((tensor_d0 & 0xFFFFu) << 16);             // bar_addr=0 | dim0 lo16
    g1[2] = (int)(((tensor_d0 >> 16) & 0xFFFFu) | ((tensor_d1 & 0xFFFFu) << 16));
    g1[3] = (int)(((tensor_d1 >> 16) & 0xFFFFu) | ((tile_d0 & 0xFFFFu) << 16));
    g1[4] = (int)(tile_d1 & 0xFFFFu);                       // tile_dim1 | tile_dim2=0
    g1[5] = (int)stride_elems;                              // dim0_stride[31:0]
    g1[6] = 0;                                              // dim0_stride hi | dim1_stride lo
    g1[7] = 0;
    i32x4 z4 = {0, 0, 0, 0};
#if __clang_major__ >= 23
    i32x8 z8 = {0, 0, 0, 0, 0, 0, 0, 0};
    __builtin_amdgcn_tensor_load_to_lds(g0, g1, z4, z4, z8, 0);
#else
    __builtin_amdgcn_tensor_load_to_lds(g0, g1, z4, z4, 0);
#endif
}
#endif

// --------------------------- conversion kernels ----------------------------

__global__ void conv_bf16_kernel(const float* __restrict__ in,
                                 unsigned short* __restrict__ out, int n) {
    int i = blockIdx.x * 256 + threadIdx.x;
    if (i < n) out[i] = f2bf(in[i]);
}

// W [1024][4096] f32 row-major  ->  WT [4096][1024] bf16 (n-major)
__global__ void conv_transpose_kernel(const float* __restrict__ in,
                                      unsigned short* __restrict__ out) {
    int i = blockIdx.x * 256 + threadIdx.x;   // 0 .. 4194303
    int k = i >> 12;                          // / 4096
    int n = i & 4095;
    out[(size_t)n * HH + k] = f2bf(in[i]);
}

__global__ void init_state_kernel(unsigned short* hb0, unsigned short* hb1,
                                  float* hf, float* cf) {
    int i = blockIdx.x * 256 + threadIdx.x;   // 32768 total
    hb0[i] = 0; hb1[i] = 0; hf[i] = 0.0f; cf[i] = 0.0f;
}

// --------------------- big GEMM: xproj = A @ W^T + bias --------------------
// A    : [8192][1024] bf16 row-major
// WT   : [4096][1024] bf16 (n-major)
// C    : [8192][4096] bf16
// grid : 2048 blocks (64 M x 32 N), 256 threads = 8 waves
// block tile 128x128, wave tile 16x128, K-step 32. A tile via TDM, double-buffered.

__global__ __launch_bounds__(256) void gemm_xproj_kernel(
    const unsigned short* __restrict__ A,
    const unsigned short* __restrict__ WT,
    const float* __restrict__ bias,
    unsigned short* __restrict__ C)
{
    __shared__ unsigned short sA[2][128 * 32];   // 2 x 8KB ping-pong

    const int tid  = threadIdx.x;
    const int wave = tid >> 5;
    const int lane = tid & 31;
    const int hi   = lane >> 4;
    const int lm   = lane & 15;

    const int bm = blockIdx.x & 63;
    const int bn = blockIdx.x >> 6;
    const int m0 = bm * 128;
    const int n0 = bn * 128;

    v8f acc[8];
    #pragma unroll
    for (int i = 0; i < 8; ++i) acc[i] = (v8f){0.f,0.f,0.f,0.f,0.f,0.f,0.f,0.f};

#if USE_TDM
    if (tid < 32) {   // wave 0: prime first A tile (128x32) into buffer 0
        tdm_load_2d(0u, A + (size_t)m0 * HH, 32, 128, HH, ROWS, HH);
    }
#endif

    for (int k0 = 0, it = 0; k0 < HH; k0 += 32, ++it) {
        const int buf = it & 1;
#if USE_TDM
        if (tid < 32) {
            if (k0 + 32 < HH) {       // issue next tile, then wait for current
                tdm_load_2d((unsigned)((buf ^ 1) * 128 * 32 * 2),
                            A + (size_t)m0 * HH + (k0 + 32), 32, 128, HH, ROWS, HH);
                __builtin_amdgcn_s_wait_tensorcnt(1);
            } else {
                __builtin_amdgcn_s_wait_tensorcnt(0);
            }
        }
#else
        {   // manual staging fallback
            int row = tid >> 1;
            int kk  = (tid & 1) * 16;
            const uint4* src = (const uint4*)(A + (size_t)(m0 + row) * HH + k0 + kk);
            uint4* dst = (uint4*)(&sA[buf][row * 32 + kk]);
            dst[0] = src[0];
            dst[1] = src[1];
        }
#endif
        __syncthreads();
        lds_publish(&sA[buf][0]);   // TDM wrote this buffer behind the compiler's back

        // A fragment (contiguous K order, matching B: lane<16 K 0..15; lane>=16 K 16..31)
        ABf a;
        {
            const unsigned short* p = &sA[buf][(16 * wave + lm) * 32] + (hi ? 16 : 0);
            a.q[0] = *(const uint4*)(p);
            a.q[1] = *(const uint4*)(p + 8);
        }
        #pragma unroll
        for (int nt = 0; nt < 8; ++nt) {
            // B fragment: lane<16: K=0..15, lane>=16: K=16..31, column = lane&15
            ABf b;
            const unsigned short* p = WT + (size_t)(n0 + nt * 16 + lm) * HH + k0 + (hi ? 16 : 0);
            b.q[0] = ((const uint4*)p)[0];
            b.q[1] = ((const uint4*)p)[1];
            acc[nt] = __builtin_amdgcn_wmma_f32_16x16x32_bf16(
                false, a.v, false, b.v, (short)0, acc[nt], false, false);
        }
        __syncthreads();
    }

    // epilogue: C/D layout  M = v + 8*hi, N = lane&15
    #pragma unroll
    for (int nt = 0; nt < 8; ++nt) {
        #pragma unroll
        for (int v = 0; v < 8; ++v) {
            int row = m0 + 16 * wave + v + 8 * hi;
            int col = n0 + nt * 16 + lm;
            float val = acc[nt][v] + bias[col];
            C[(size_t)row * G4H + col] = f2bf(val);
        }
    }
}

// ------------------------ fused recurrent step kernel ----------------------
// z = xproj[t] + h @ Wh ; gates ; state update ; masked outputs.
// grid: 8 blocks x 256 threads (8 waves). wave owns hidden cols u0..u0+15,
// computing all 4 gate tiles (i,j,f,o) for both batch tiles (b 0-15, 16-31).
// Whole 32x1024 bf16 h-state staged into LDS once per block via TDM.

__global__ __launch_bounds__(256) void lstm_step_kernel(
    const unsigned short* __restrict__ xproj,   // [8192][4096] bf16, row = b*T+t
    const unsigned short* __restrict__ WhT,     // [4096][1024] bf16, n-major
    const unsigned short* __restrict__ h_in,    // [32][1024] bf16 (prev h)
    unsigned short* __restrict__ h_out,         // [32][1024] bf16 (new h)
    float* __restrict__ h_f32,                  // [32][1024] carried h (f32)
    float* __restrict__ c_f32,                  // [32][1024] carried c (f32)
    const int* __restrict__ lengths,            // [32]
    unsigned short* __restrict__ y_bf16,        // layer0 output (feeds next layer) or null
    float* __restrict__ y_f32,                  // layer1 output (d_out rnnout) or null
    float* __restrict__ d_final,                // d_out base (cs/hs written at t==T-1)
    int t, int layer)
{
    const int tid  = threadIdx.x;
    const int wave = tid >> 5;
    const int lane = tid & 31;
    const int hi   = lane >> 4;
    const int lm   = lane & 15;
    const int u0   = blockIdx.x * 128 + wave * 16;

#if USE_TDM
    __shared__ unsigned short sH[BATCH * HH];   // 64 KB: full prev-h tile
    if (tid < 32) {
        tdm_load_2d(0u, h_in, HH, BATCH, HH, BATCH, HH);
        __builtin_amdgcn_s_wait_tensorcnt(0);
    }
    __syncthreads();
    lds_publish(sH);                            // TDM wrote sH behind the compiler's back
    const unsigned short* __restrict__ hsrc = sH;
#else
    const unsigned short* __restrict__ hsrc = h_in;
#endif

    v8f acc[4][2];
    #pragma unroll
    for (int g = 0; g < 4; ++g)
        #pragma unroll
        for (int mt = 0; mt < 2; ++mt)
            acc[g][mt] = (v8f){0.f,0.f,0.f,0.f,0.f,0.f,0.f,0.f};

    for (int k0 = 0; k0 < HH; k0 += 32) {
        ABf a[2];
        #pragma unroll
        for (int mt = 0; mt < 2; ++mt) {
            const unsigned short* p = hsrc + (size_t)(mt * 16 + lm) * HH + k0 + (hi ? 16 : 0);
            a[mt].q[0] = *(const uint4*)(p);
            a[mt].q[1] = *(const uint4*)(p + 8);
        }
        #pragma unroll
        for (int g = 0; g < 4; ++g) {
            ABf b;
            const unsigned short* p = WhT + (size_t)(g * HH + u0 + lm) * HH + k0 + (hi ? 16 : 0);
            b.q[0] = ((const uint4*)p)[0];
            b.q[1] = ((const uint4*)p)[1];
            #pragma unroll
            for (int mt = 0; mt < 2; ++mt) {
                acc[g][mt] = __builtin_amdgcn_wmma_f32_16x16x32_bf16(
                    false, a[mt].v, false, b.v, (short)0, acc[g][mt], false, false);
            }
        }
    }

    // gates + state update. element (mt, v): b = mt*16 + v + 8*hi, u = u0 + lm
    const int u = u0 + lm;
    #pragma unroll
    for (int mt = 0; mt < 2; ++mt) {
        #pragma unroll
        for (int v = 0; v < 8; ++v) {
            const int b = mt * 16 + v + 8 * hi;
            const size_t r  = (size_t)b * TT + t;       // xproj row
            const size_t su = (size_t)b * HH + u;       // state index
            const unsigned short* xp = xproj + r * G4H + u;

            float zi = acc[0][mt][v] + bf2f(xp[0]);
            float zj = acc[1][mt][v] + bf2f(xp[HH]);
            float zf = acc[2][mt][v] + bf2f(xp[2 * HH]);
            float zo = acc[3][mt][v] + bf2f(xp[3 * HH]);

            float cp = c_f32[su];
            float cn = sigmoidf(zf + 1.0f) * cp + sigmoidf(zi) * tanhf(zj);
            float hn = sigmoidf(zo) * tanhf(cn);

            bool  m  = (t < lengths[b]);
            float hp = h_f32[su];
            float h2 = m ? hn : hp;
            float c2 = m ? cn : cp;

            c_f32[su] = c2;
            h_f32[su] = h2;
            h_out[su] = f2bf(h2);

            float outv = m ? hn : 0.0f;
            if (layer == 0) y_bf16[r * HH + u] = f2bf(outv);
            else            y_f32[r * HH + u]  = outv;

            if (t == TT - 1) {
                d_final[CS_OFF + layer * (BATCH * HH) + su] = c2;
                d_final[HS_OFF + layer * (BATCH * HH) + su] = h2;
            }
        }
    }
}

// ------------------------------- launcher ----------------------------------

extern "C" void kernel_launch(void* const* d_in, const int* in_sizes, int n_in,
                              void* d_out, int out_size, void* d_ws, size_t ws_size,
                              hipStream_t stream) {
    const float* x       = (const float*)d_in[0];   // [32][256][1024]
    const int*   lengths = (const int*)  d_in[1];   // [32]
    const float* Wx      = (const float*)d_in[2];   // [2][1024][4096]
    const float* Wh      = (const float*)d_in[3];   // [2][1024][4096]
    const float* bias    = (const float*)d_in[4];   // [2][4096]
    float*       out     = (float*)d_out;

    char* ws = (char*)d_ws;
    size_t o = 0;
    unsigned short* xin   = (unsigned short*)(ws + o); o += (size_t)ROWS * HH * 2;      // 16 MiB (reused as y0)
    unsigned short* WxT   = (unsigned short*)(ws + o); o += (size_t)2 * G4H * HH * 2;   // 16 MiB
    unsigned short* WhT   = (unsigned short*)(ws + o); o += (size_t)2 * G4H * HH * 2;   // 16 MiB
    unsigned short* xproj = (unsigned short*)(ws + o); o += (size_t)ROWS * G4H * 2;     // 64 MiB
    unsigned short* hb0   = (unsigned short*)(ws + o); o += (size_t)BATCH * HH * 2;
    unsigned short* hb1   = (unsigned short*)(ws + o); o += (size_t)BATCH * HH * 2;
    float*          hf    = (float*)(ws + o);          o += (size_t)BATCH * HH * 4;
    float*          cf    = (float*)(ws + o);          o += (size_t)BATCH * HH * 4;
    (void)ws_size; (void)in_sizes; (void)n_in; (void)out_size;

    const size_t WELEM = (size_t)HH * G4H;   // 4194304 per layer

    // x -> bf16
    conv_bf16_kernel<<<(ROWS * HH) / 256, 256, 0, stream>>>(x, xin, ROWS * HH);
    // weights -> bf16, transposed to [n][k]
    for (int l = 0; l < 2; ++l) {
        conv_transpose_kernel<<<WELEM / 256, 256, 0, stream>>>(Wx + l * WELEM, WxT + l * WELEM);
        conv_transpose_kernel<<<WELEM / 256, 256, 0, stream>>>(Wh + l * WELEM, WhT + l * WELEM);
    }

    for (int l = 0; l < 2; ++l) {
        // xproj = input @ Wx[l] + b[l]   (input = x for l=0, masked layer0 out for l=1)
        gemm_xproj_kernel<<<2048, 256, 0, stream>>>(xin, WxT + l * WELEM,
                                                    bias + l * G4H, xproj);
        init_state_kernel<<<(BATCH * HH) / 256, 256, 0, stream>>>(hb0, hb1, hf, cf);

        for (int t = 0; t < TT; ++t) {
            unsigned short* hin  = (t & 1) ? hb1 : hb0;
            unsigned short* hout = (t & 1) ? hb0 : hb1;
            lstm_step_kernel<<<8, 256, 0, stream>>>(
                xproj, WhT + l * WELEM, hin, hout, hf, cf, lengths,
                (l == 0) ? xin : nullptr,     // layer0 output feeds layer1 (bf16)
                (l == 1) ? out : nullptr,     // layer1 output -> d_out rnnout (f32)
                out, t, l);
        }
    }
}
